// Decoder_9955734192537
// MI455X (gfx1250) — matrix operand order repaired and checked by reading, not compiled, not run on hardware
//
#include <hip/hip_runtime.h>
#include <stdint.h>

// ---------------------------------------------------------------------------
// Persistent LSTM for MI455X (gfx1250).
//   grid = 8 batch-groups (16 rows) x 16 hidden-slices = 128 workgroups.
//   Each WG keeps its 128x512 f16 slice of W_hh in LDS for the whole
//   recurrence (zero steady-state weight traffic), runs 512 time steps with
//   v_wmma_f32_16x16x32_f16, and exchanges h slices with its 15 sibling WGs
//   through an L2-resident buffer + two-phase atomic barrier per step.
//   Phase-2 (readers-done) is split arrive/wait so its latency hides behind
//   the next step's WMMA GEMM.
// ---------------------------------------------------------------------------

typedef __attribute__((ext_vector_type(16))) _Float16 v16h;
typedef __attribute__((ext_vector_type(8)))  _Float16 v8h;
typedef __attribute__((ext_vector_type(8)))  float    v8f;

#define F_IN    128
#define HID     512
#define T_STEPS 512
#define M_ROWS  16
#define NSLICE  16          // workgroups per batch group
#define THREADS 256         // 8 waves (wave32)
#define KP      520         // LDS pitch (halfs) for W slice: 260 DW -> bank step 4
#define KBLK    16          // 512 / 32 k-steps

#define USE_ASYNC_REBUILD 1

__device__ __forceinline__ float sigmoid_f(float x) {
  return 1.0f / (1.0f + __expf(-x));          // exp(+inf) handled -> 0
}
__device__ __forceinline__ float tanh_f(float x) {
  x = fminf(fmaxf(x, -15.0f), 15.0f);         // avoid inf/inf
  float e = __expf(2.0f * x);
  return (e - 1.0f) / (e + 1.0f);
}

__global__ void init_ws_kernel(unsigned int* cnt) {
  int i = blockIdx.x * blockDim.x + threadIdx.x;
  if (i < 1024) cnt[i] = 0u;
}

__global__ __launch_bounds__(THREADS)
void lstm_persistent(const float* __restrict__ Z,
                     const int*   __restrict__ seq_len,
                     const float* __restrict__ W_ih,
                     const float* __restrict__ W_hh,
                     const float* __restrict__ b_ih,
                     const float* __restrict__ b_hh,
                     float*       __restrict__ out,
                     unsigned int* __restrict__ cnt_base,
                     _Float16*    __restrict__ hx_base)
{
  // ~158 KB LDS total -> 2 workgroups per WGP (320 KB).
  __shared__ __align__(32) _Float16 Wlds[128 * KP];        // 133,120 B
  __shared__ __align__(32) _Float16 hsw[KBLK * 32 * 16];   //  16,384 B (A-frag layout)
  __shared__ __align__(16) float    gbuf[8 * 256];         //   8,192 B

  const int tid   = threadIdx.x;
  const int wave  = tid >> 5;
  const int lane  = tid & 31;
  const int wg    = blockIdx.x;
  const int group = wg >> 4;          // batch tile
  const int slice = wg & 15;          // hidden slice (32 cols)
  const int b0    = group * M_ROWS;

  unsigned int* cnt = cnt_base + group * 32;               // 128B-padded counter
  _Float16* hx = hx_base + (size_t)group * (M_ROWS * HID); // [16][512] f16 exchange

  // ---- one-time: W_hh slice (f32 -> f16) into LDS, row-major pitch KP ----
  for (int idx = tid; idx < 128 * HID; idx += THREADS) {
    int ln = idx >> 9;                 // local gate row 0..127
    int k  = idx & (HID - 1);
    int ng = ((ln >> 5) * HID) + slice * 32 + (ln & 31);   // global gate row
    Wlds[ln * KP + k] = (_Float16)W_hh[(size_t)ng * HID + k];
  }
  for (int idx = tid; idx < KBLK * 32 * 16; idx += THREADS)
    hsw[idx] = (_Float16)0.0f;                             // h0 = 0

  // ---- one-time: fp32 x_proj fragment for this wave's 16x16 gate tile ----
  const int blk  = wave >> 1;          // gate block: 0=i 1=f 2=g 3=o
  const int wsub = wave & 1;           // which 16-col half of the 32-col slice
  const int hi   = lane >> 4;
  const int lcol = lane & 15;
  const int ncol = blk * HID + slice * 32 + wsub * 16 + lcol;  // global gate col
  v8f xp;
  {
    float bias = b_ih[ncol] + b_hh[ncol];
    const float* wr = W_ih + (size_t)ncol * F_IN;
    #pragma unroll 1
    for (int r = 0; r < 8; ++r) {
      int m = r + 8 * hi;
      const float* zr = Z + (size_t)(b0 + m) * F_IN;
      float s = bias;
      for (int f = 0; f < F_IN; ++f) s = fmaf(zr[f], wr[f], s);
      xp[r] = s;
    }
  }

  // ---- per-thread cell state: 2 hidden elements (16 rows x 32 cols = 512) ----
  const int e1   = tid + 256;
  const int sub0 = 0,         idx0 = tid,       row0 = idx0 >> 4, col0 = idx0 & 15;
  const int sub1 = e1 >> 8,   idx1 = e1 & 255,  row1 = idx1 >> 4, col1 = idx1 & 15;
  const int hc0  = slice * 32 + sub0 * 16 + col0;
  const int hc1  = slice * 32 + sub1 * 16 + col1;
  const int L0   = seq_len[b0 + row0];
  const int L1   = seq_len[b0 + row1];
  float c0 = 0.0f, c1 = 0.0f;

  const _Float16* wbase = Wlds + (size_t)(blk * 32 + wsub * 16 + lcol) * KP;
  __syncthreads();

  #pragma unroll 1
  for (int t = 0; t < T_STEPS; ++t) {
    // ================= GEMM: gates = x_proj + h @ W_hh^T =================
    v8f acc = xp;
    #pragma unroll
    for (int kb = 0; kb < KBLK; ++kb) {
      v16h a = *(const v16h*)(hsw + (kb * 32 + lane) * 16);      // ISA A layout
      union { v16h v; v8h p[2]; } bu;
      const _Float16* bp = wbase + kb * 32 + hi * 8;             // k 8h..+8, 16+8h..+8
      bu.p[0] = *(const v8h*)(bp);
      bu.p[1] = *(const v8h*)(bp + 16);
      acc = __builtin_amdgcn_wmma_f32_16x16x32_f16(
          false, a, false, bu.v, (short)0, acc, false, false);
    }
    #pragma unroll
    for (int r = 0; r < 8; ++r)
      gbuf[wave * 256 + (r + 8 * hi) * 16 + lcol] = acc[r];
    __syncthreads();

    // ---- deferred phase-2 wait of step t-1: all sibling WGs must have
    //      finished READING h(t-1) before we overwrite hx with h(t).
    //      (Latency hidden behind the GEMM above.)
    if (tid == 0 && t > 0) {
      unsigned tgt = (unsigned)(NSLICE * 2 * t);
      while (__hip_atomic_load(cnt, __ATOMIC_ACQUIRE, __HIP_MEMORY_SCOPE_AGENT) < tgt)
        __builtin_amdgcn_s_sleep(1);
    }
    __syncthreads();

    // ================= elementwise LSTM cell =================
    {
      float i = sigmoid_f(gbuf[(0 + sub0) * 256 + idx0]);
      float f = sigmoid_f(gbuf[(2 + sub0) * 256 + idx0]);
      float g = tanh_f  (gbuf[(4 + sub0) * 256 + idx0]);
      float o = sigmoid_f(gbuf[(6 + sub0) * 256 + idx0]);
      c0 = f * c0 + i * g;
      float h = o * tanh_f(c0);
      hx[row0 * HID + hc0] = (_Float16)h;
      out[(size_t)(b0 + row0) * T_STEPS * HID + (size_t)t * HID + hc0] =
          (t < L0) ? h : 0.0f;
    }
    {
      float i = sigmoid_f(gbuf[(0 + sub1) * 256 + idx1]);
      float f = sigmoid_f(gbuf[(2 + sub1) * 256 + idx1]);
      float g = tanh_f  (gbuf[(4 + sub1) * 256 + idx1]);
      float o = sigmoid_f(gbuf[(6 + sub1) * 256 + idx1]);
      c1 = f * c1 + i * g;
      float h = o * tanh_f(c1);
      hx[row1 * HID + hc1] = (_Float16)h;
      out[(size_t)(b0 + row1) * T_STEPS * HID + (size_t)t * HID + hc1] =
          (t < L1) ? h : 0.0f;
    }

    // ---- phase 1: all 16 WGs of this group have written their h(t) slice ----
    __threadfence();
    __syncthreads();
    if (tid == 0) {
      __hip_atomic_fetch_add(cnt, 1u, __ATOMIC_RELEASE, __HIP_MEMORY_SCOPE_AGENT);
      unsigned tgt = (unsigned)(NSLICE * (2 * t + 1));
      while (__hip_atomic_load(cnt, __ATOMIC_ACQUIRE, __HIP_MEMORY_SCOPE_AGENT) < tgt)
        __builtin_amdgcn_s_sleep(1);
    }
    __syncthreads();
    __threadfence();   // acquire: invalidate stale near caches before reading hx

    // ---- rebuild swizzled f16 h (full 16x512) in LDS from exchange buffer ----
#if USE_ASYNC_REBUILD
    for (int s = tid; s < KBLK * 32; s += THREADS) {
      int kb = s >> 5, l = s & 31, m = l & 15, khi = (l >> 4) * 8;
      const _Float16* src0 = hx + m * HID + kb * 32 + khi;   // k in [kb*32+khi, +8)
      const _Float16* src1 = src0 + 16;                       // k+16 half
      uint32_t d0 = (uint32_t)(uint64_t)(const void*)(hsw + s * 16);
      asm volatile("global_load_async_to_lds_b128 %0, %1, off scope:SCOPE_DEV"
                   :: "v"(d0), "v"(src0) : "memory");
      uint32_t d1 = d0 + 16;
      asm volatile("global_load_async_to_lds_b128 %0, %1, off scope:SCOPE_DEV"
                   :: "v"(d1), "v"(src1) : "memory");
    }
    asm volatile("s_wait_asynccnt 0x0" ::: "memory");
#else
    for (int s = tid; s < KBLK * 32; s += THREADS) {
      int kb = s >> 5, l = s & 31, m = l & 15, khi = (l >> 4) * 8;
      const uint4* src0 = (const uint4*)(hx + m * HID + kb * 32 + khi);
      const uint4* src1 = (const uint4*)(hx + m * HID + kb * 32 + khi + 16);
      uint4* dst = (uint4*)(hsw + s * 16);
      dst[0] = *src0;
      dst[1] = *src1;
    }
#endif
    __syncthreads();   // hsw complete; also orders rebuild reads before arrive

    // ---- phase 2 ARRIVE only (readers of h(t) done); the matching wait is
    //      deferred into the next iteration, hidden behind its GEMM.
    if (tid == 0)
      __hip_atomic_fetch_add(cnt, 1u, __ATOMIC_ACQ_REL, __HIP_MEMORY_SCOPE_AGENT);
  }
}

extern "C" void kernel_launch(void* const* d_in, const int* in_sizes, int n_in,
                              void* d_out, int out_size, void* d_ws, size_t ws_size,
                              hipStream_t stream) {
  const float* Z       = (const float*)d_in[0];
  const int*   seq_len = (const int*)  d_in[1];
  const float* W_ih    = (const float*)d_in[2];
  const float* W_hh    = (const float*)d_in[3];
  const float* b_ih    = (const float*)d_in[4];
  const float* b_hh    = (const float*)d_in[5];
  float* out = (float*)d_out;

  unsigned int* cnt = (unsigned int*)d_ws;                 // 4 KB counter region
  _Float16*     hx  = (_Float16*)((char*)d_ws + 4096);     // 8 x 16 x 512 f16

  init_ws_kernel<<<4, 256, 0, stream>>>(cnt);
  lstm_persistent<<<8 * NSLICE, THREADS, 0, stream>>>(
      Z, seq_len, W_ih, W_hh, b_ih, b_hh, out, cnt, hx);
}
